// NMS_12764642804265
// MI455X (gfx1250) — matrix operand order repaired and checked by reading, not compiled, not run on hardware
//
#include <hip/hip_runtime.h>

#define BATCH_N 25200
#define NSORT   32768          // next pow2 >= 25200, fully sorted in LDS (256KB of the 320KB WGP LDS)
#define TOPK    2048
#define MAXOUT  50
#define CONF_T  0.25f
#define IOU_T   0.45f
#define NEGV    -1000000000.0f

typedef __attribute__((ext_vector_type(16))) _Float16 v16h;
typedef __attribute__((ext_vector_type(8)))  float    v8f;

// ---------------------------------------------------------------------------
// Kernel 1: per-batch score threshold + full bitonic sort of 32768 (score,idx)
// pairs in LDS, emit sorted top-2048 candidate rows into workspace.
// Candidate layout: 8 f32 per row: x1,y1,x2,y2,score,cls,area,valid
// ---------------------------------------------------------------------------
__global__ __launch_bounds__(1024) void topk_sort_kernel(const float* __restrict__ x,
                                                         float* __restrict__ cand) {
  extern __shared__ char smem[];
  float* ss = (float*)smem;                         // NSORT scores
  int*   si = (int*)(smem + NSORT * sizeof(float)); // NSORT indices
  const int b   = blockIdx.x;
  const int tid = threadIdx.x;
  const float* xb = x + (size_t)b * BATCH_N * 6;

  for (int t = tid; t < NSORT; t += blockDim.x) {
    float s = NEGV;
    if (t < BATCH_N) {
      float c = xb[t * 6 + 4];
      s = (c > CONF_T) ? c : NEGV;
    }
    ss[t] = s;
    si[t] = t;
  }
  __syncthreads();

  // Bitonic sort, descending. Each thread owns disjoint (t, t^j) pairs.
  for (unsigned k = 2; k <= NSORT; k <<= 1) {
    for (unsigned j = k >> 1; j > 0; j >>= 1) {
      for (unsigned t = tid; t < NSORT; t += blockDim.x) {
        unsigned p = t ^ j;
        if (p > t) {
          float a = ss[t], c = ss[p];
          bool desc = ((t & k) == 0);
          if (desc ? (a < c) : (a > c)) {
            ss[t] = c; ss[p] = a;
            int ia = si[t]; si[t] = si[p]; si[p] = ia;
          }
        }
      }
      __syncthreads();
    }
  }

  for (int t = tid; t < TOPK; t += blockDim.x) {
    float sc = ss[t];
    int  idx = si[t];
    float x1 = -1.f, y1 = -1.f, x2 = -1.f, y2 = -1.f, cls = -1.f, valid = 0.f;
    if (sc > 0.5f * NEGV && idx < BATCH_N) {
      const float* r = xb + (size_t)idx * 6;
      x1 = r[0]; y1 = r[1]; x2 = r[2]; y2 = r[3]; cls = r[5]; valid = 1.f;
    } else {
      sc = NEGV;
    }
    float area = fmaxf(x2 - x1, 0.f) * fmaxf(y2 - y1, 0.f);
    float* o = cand + ((size_t)b * TOPK + t) * 8;
    o[0] = x1; o[1] = y1; o[2] = x2; o[3] = y2;
    o[4] = sc; o[5] = cls; o[6] = area; o[7] = valid;
  }
}

// Division-free IoU threshold test: iou >= t  <=>  inter >= t * max(union,eps)
__device__ __forceinline__ bool iou_ge(float ax1, float ay1, float ax2, float ay2, float aa,
                                       float bx1, float by1, float bx2, float by2, float ba) {
  float iw = fmaxf(fminf(ax2, bx2) - fmaxf(ax1, bx1), 0.f);
  float ih = fmaxf(fminf(ay2, by2) - fmaxf(ay1, by1), 0.f);
  float inter = iw * ih;
  float uni = fmaxf(aa + ba - inter, 1e-9f);
  return inter >= IOU_T * uni;
}

// ---------------------------------------------------------------------------
// Kernel 2: block-parallel NMS. Serial depth cut from 2048 to 64 block-steps.
// Per 32-row block: wave0 does the exact serial scan (wave32: 1 lane per row),
// then all 16 waves propagate suppression to the columns via
// v_wmma_f32_16x16x32_f16: D = A(keep flags, replicated rows) x B(IoU>=thr
// indicator tile). Survivors compacted (first 50, already score-sorted).
// ---------------------------------------------------------------------------
__global__ __launch_bounds__(512) void nms_wmma_kernel(const float* __restrict__ cand,
                                                       float* __restrict__ out) {
  extern __shared__ char smem[];
  float*    cb        = (float*)smem;              // TOPK*8 candidate rows
  float*    suppr     = cb + TOPK * 8;             // TOPK suppression flags (0/1)
  unsigned* keepmasks = (unsigned*)(suppr + TOPK); // TOPK/32 keep bitmasks

  const int b    = blockIdx.x;
  const int tid  = threadIdx.x;
  const int lane = tid & 31;
  const int wave = tid >> 5;

  float* ob = out + (size_t)b * MAXOUT * 6;
  for (int t = tid; t < MAXOUT * 6; t += blockDim.x) ob[t] = -1.0f;

  const float* cin = cand + (size_t)b * TOPK * 8;
  for (int t = tid; t < TOPK * 8; t += blockDim.x) cb[t] = cin[t];
  for (int t = tid; t < TOPK; t += blockDim.x) suppr[t] = 0.f;
  __syncthreads();

  const int NBLK = TOPK / 32; // 64
  for (int blk = 0; blk < NBLK; ++blk) {
    const int rb = blk * 32;

    // --- serial intra-block scan on wave 0 (exact reference semantics) ---
    if (wave == 0) {
      const float* mb = cb + (size_t)(rb + lane) * 8;
      float bx1 = mb[0], by1 = mb[1], bx2 = mb[2], by2 = mb[3];
      float barea = mb[6], bval = mb[7];
      int supL = 0;
      unsigned keepm = 0;
      for (int i = 0; i < 32; ++i) {
        float ax1 = __shfl(bx1, i, 32), ay1 = __shfl(by1, i, 32);
        float ax2 = __shfl(bx2, i, 32), ay2 = __shfl(by2, i, 32);
        float aarea = __shfl(barea, i, 32), aval = __shfl(bval, i, 32);
        int   asup = __shfl(supL, i, 32);
        bool keep_i = (aval > 0.5f) && (suppr[rb + i] < 0.5f) && (asup == 0);
        if (keep_i) { // lane-uniform
          keepm |= (1u << i);
          bool hit = iou_ge(ax1, ay1, ax2, ay2, aarea, bx1, by1, bx2, by2, barea);
          if (lane > i && hit) supL = 1;
        }
      }
      if (supL) suppr[rb + lane] = 1.0f;
      if (lane == 0) keepmasks[blk] = keepm;
    }
    __syncthreads();

    // --- WMMA propagation: suppr[col] |= (sum_k keep[k] * ind[k][col]) > 0 ---
    const unsigned keepm = keepmasks[blk]; // block-uniform
    if (keepm != 0) {
      // A: 16x32 f16, every output row identical = keep flags of the 32 rows.
      // A k order used here (contiguous, same as B): lanes0-15 e0..15->K0..15;
      //                  lanes16-31 e0..15->K16..31.
      v16h a;
#pragma unroll
      for (int e = 0; e < 16; ++e) {
        int bit = e + ((lane >= 16) ? 16 : 0);
        a[e] = (_Float16)((keepm >> bit) & 1u);
      }
      const int colLane = lane & 15;
      const int kbase   = (lane < 16) ? 0 : 16; // B 32x16: lanes0-15 K=0..15, lanes16-31 K=16..31

      // Hoist this lane's 16 K-row boxes into registers (invariant over tiles;
      // the suppr[] store aliases smem so the compiler cannot do this itself).
      float rx1[16], ry1[16], rx2[16], ry2[16], rar[16];
#pragma unroll
      for (int e = 0; e < 16; ++e) {
        const float* rx = cb + (size_t)(rb + kbase + e) * 8;
        rx1[e] = rx[0]; ry1[e] = rx[1]; rx2[e] = rx[2]; ry2[e] = rx[3]; rar[e] = rx[6];
      }

      for (int ct = wave; ct < TOPK / 16; ct += 16) { // wave-uniform control flow
        if (ct * 16 + 15 <= rb) continue; // all cols <= first block row: tile is zero
        const int col = ct * 16 + colLane;
        const float* cbox = cb + (size_t)col * 8;
        float cx1 = cbox[0], cy1 = cbox[1], cx2 = cbox[2], cy2 = cbox[3], carea = cbox[6];
        v16h bm;
#pragma unroll
        for (int e = 0; e < 16; ++e) {
          int r = rb + kbase + e;
          bool ind = (col > r) && iou_ge(rx1[e], ry1[e], rx2[e], ry2[e], rar[e],
                                         cx1, cy1, cx2, cy2, carea);
          bm[e] = (_Float16)(ind ? 1.0f : 0.0f);
        }
        v8f c = {};
        v8f d = __builtin_amdgcn_wmma_f32_16x16x32_f16(false, a, false, bm,
                                                       (short)0, c, false, false);
        float cnt = d[0]; // all output rows identical
        if (lane < 16 && cnt > 0.5f) suppr[col] = 1.0f;
      }
    }
    __syncthreads();
  }

  // --- first 50 survivors (candidates are score-sorted => these are top-50) ---
  if (tid < 32) {
    int outc = 0;
    for (int blk = 0; blk < NBLK && outc < MAXOUT; ++blk) {
      unsigned km = keepmasks[blk];
      bool k = (km >> lane) & 1u;
      unsigned below = (lane == 0) ? 0u : (km & ((1u << lane) - 1u));
      int pos = outc + __popc(below);
      if (k && pos < MAXOUT) {
        const float* r = cb + (size_t)(blk * 32 + lane) * 8;
        ob[pos * 6 + 0] = r[0]; ob[pos * 6 + 1] = r[1];
        ob[pos * 6 + 2] = r[2]; ob[pos * 6 + 3] = r[3];
        ob[pos * 6 + 4] = r[4]; ob[pos * 6 + 5] = r[5];
      }
      outc += __popc(km);
    }
  }
}

extern "C" void kernel_launch(void* const* d_in, const int* in_sizes, int n_in,
                              void* d_out, int out_size, void* d_ws, size_t ws_size,
                              hipStream_t stream) {
  const float* x = (const float*)d_in[0];
  float* out = (float*)d_out;
  const int B = in_sizes[0] / (BATCH_N * 6); // 16
  float* cand = (float*)d_ws;                // B * TOPK * 8 f32 = 1 MB

  const size_t lds1 = (size_t)NSORT * 8;                               // 256 KB
  const size_t lds2 = ((size_t)TOPK * 8 + TOPK) * 4 + (TOPK / 32) * 4; // ~72.25 KB
  hipFuncSetAttribute((const void*)topk_sort_kernel,
                      hipFuncAttributeMaxDynamicSharedMemorySize, (int)lds1);
  hipFuncSetAttribute((const void*)nms_wmma_kernel,
                      hipFuncAttributeMaxDynamicSharedMemorySize, (int)lds2);

  topk_sort_kernel<<<B, 1024, lds1, stream>>>(x, cand);
  nms_wmma_kernel<<<B, 512, lds2, stream>>>(cand, out);
}
